// CRF_88768384073873
// MI455X (gfx1250) — compile-verified, matrix operand order checked by reading, not run on hardware
//
#include <hip/hip_runtime.h>
#include <hip/hip_bf16.h>

#define SEQ   256
#define BATCH 64
#define NTAG  128
#define TB    16      // batches per workgroup == WMMA M
#define ESTR  136     // ET row stride (f16), mult of 8 -> 16B aligned rows, bank-spread
#define ASTR  136     // A row stride (f16)
#define SSTR  164     // score row stride (f32): 164*4 % 16 == 0, good bank separation

#define LOG2E 1.44269504088896340736f
#define LN2   0.69314718055994530942f

// raw hardware transcendentals: v_exp_f32 (2^x), v_log_f32 (log2 x)
#define EXP2F(x) __builtin_amdgcn_exp2f(x)
#define LOG2F(x) __builtin_amdgcn_logf(x)

typedef __attribute__((ext_vector_type(16))) _Float16 v16h;
typedef __attribute__((ext_vector_type(8)))  _Float16 v8h;
typedef __attribute__((ext_vector_type(8)))  float    v8f;

// xor-reductions over the 16-lane half-wave groups (xor masks 1,2,4,8 never cross bit 4)
__device__ __forceinline__ float swz_max16(float v) {
  v = fmaxf(v, __int_as_float(__builtin_amdgcn_ds_swizzle(__float_as_int(v), 0x041F)));
  v = fmaxf(v, __int_as_float(__builtin_amdgcn_ds_swizzle(__float_as_int(v), 0x081F)));
  v = fmaxf(v, __int_as_float(__builtin_amdgcn_ds_swizzle(__float_as_int(v), 0x101F)));
  v = fmaxf(v, __int_as_float(__builtin_amdgcn_ds_swizzle(__float_as_int(v), 0x201F)));
  return v;
}
__device__ __forceinline__ float swz_sum16(float v) {
  v += __int_as_float(__builtin_amdgcn_ds_swizzle(__float_as_int(v), 0x041F));
  v += __int_as_float(__builtin_amdgcn_ds_swizzle(__float_as_int(v), 0x081F));
  v += __int_as_float(__builtin_amdgcn_ds_swizzle(__float_as_int(v), 0x101F));
  v += __int_as_float(__builtin_amdgcn_ds_swizzle(__float_as_int(v), 0x201F));
  return v;
}

// 16-bit A/B fragment (ISA 7.12.2): lane holds K = [off, off+8) in v0..3 and
// [off+16, off+24) in v4..7, where off already includes the half-wave K shift.
__device__ __forceinline__ v16h load_frag(const _Float16* rowbase, int off) {
  v8h c0 = *(const v8h*)(rowbase + off);
  v8h c1 = *(const v8h*)(rowbase + off + 16);
  return __builtin_shufflevector(c0, c1, 0,1,2,3,4,5,6,7,8,9,10,11,12,13,14,15);
}

// ---------------- normalizer (forward algorithm) ----------------
__global__ void __launch_bounds__(256)
crf_forward_kernel(const float* __restrict__ em, const unsigned char* __restrict__ mask,
                   const float* __restrict__ startT, const float* __restrict__ endT,
                   const float* __restrict__ trans, float* __restrict__ denom) {
  __shared__ __align__(16) _Float16 ET16[NTAG * ESTR];  // ET16[n][k] = exp(trans[k][n])
  __shared__ __align__(16) _Float16 A16[TB * ASTR];     // A16[b][i]  = exp(score-M) f16
  __shared__ __align__(16) float scoreS[TB * SSTR];
  __shared__ __align__(16) float Mld[TB];

  const int tid  = threadIdx.x;
  const int lane = tid & 31;
  const int wave = tid >> 5;          // 0..7
  const int j0   = lane & 15;
  const int half = lane >> 4;         // 0/1
  const int b0   = blockIdx.x * TB;

  // one-time: E^T into LDS (f16); trans ~ 0.1*N(0,1) so exp is tame
  for (int q = 0; q < 64; ++q) {
    int idx = tid + 256 * q;                       // 0..16383 over trans flat [i][j]
    int i = idx >> 7, j = idx & 127;
    ET16[j * ESTR + i] = (_Float16)EXP2F(trans[idx] * LOG2E);
  }
  // init: score[b][j] = start[j] + em[0][b0+b][j]
  for (int q = 0; q < 8; ++q) {
    int idx = tid + 256 * q;                       // 0..2047
    int b = idx >> 7, j = idx & 127;
    scoreS[b * SSTR + j] = startT[j] + em[(size_t)(b0 + b) * NTAG + j];
  }
  __syncthreads();

  const int bb    = 2 * wave + half;   // batch handled in max/exp phases
  const int cbase = 8 * j0;            // this lane's contiguous column block (phase 1/2)
  const int nbase = wave * 16;         // WMMA tile columns
  const int khalf = half * 8;          // K shift inside fragment
  const int mrow  = half * 8;          // D rows: lanes0-15 -> M=r, lanes16-31 -> M=8+r

  for (int step = 1; step < SEQ; ++step) {
    const float* em_t = em + ((size_t)step * BATCH + b0) * NTAG;
    const unsigned char* mk = mask + (size_t)step * BATCH + b0;

    // issue this step's global loads early (hidden under phase 1/2 + barrier)
    float emv[8];
#pragma unroll
    for (int r = 0; r < 8; ++r)
      emv[r] = em_t[(mrow + r) * NTAG + nbase + j0];
    const unsigned long long mw = *(const unsigned long long*)&mk[mrow];
    if (step + 1 < SEQ && tid < 64)    // 64 * 128B lines = this block's next em slice
      __builtin_prefetch(em + ((size_t)(step + 1) * BATCH + b0) * NTAG + tid * 32, 0, 0);

    // phase 1: per-batch row max (contiguous 8 columns/lane, 2x ds_load_b128)
    const float* srow = &scoreS[bb * SSTR + cbase];
    float4 s0 = *(const float4*)(srow);
    float4 s1 = *(const float4*)(srow + 4);
    float vals[8] = {s0.x, s0.y, s0.z, s0.w, s1.x, s1.y, s1.z, s1.w};
    float m = vals[0];
#pragma unroll
    for (int i = 1; i < 8; ++i) m = fmaxf(m, vals[i]);
    m = swz_max16(m);
    if (j0 == 0) Mld[bb] = m;
    // phase 2: a = exp(score - M) -> packed f16, one ds_store_b128
    v8h a;
#pragma unroll
    for (int i = 0; i < 8; ++i)
      a[i] = (_Float16)EXP2F((vals[i] - m) * LOG2E);
    *(v8h*)&A16[bb * ASTR + cbase] = a;
    __syncthreads();

    // old scores + M, loaded up-front so WMMA latency hides them (LDS is in-order
    // within a wave, so these loads complete before this wave's stores below)
    float oldv[8];
#pragma unroll
    for (int r = 0; r < 8; ++r)
      oldv[r] = scoreS[(mrow + r) * SSTR + nbase + j0];
    float4 mlo = *(const float4*)&Mld[mrow];
    float4 mhi = *(const float4*)&Mld[mrow + 4];
    float base[8] = {emv[0] + mlo.x, emv[1] + mlo.y, emv[2] + mlo.z, emv[3] + mlo.w,
                     emv[4] + mhi.x, emv[5] + mhi.y, emv[6] + mhi.z, emv[7] + mhi.w};

    // phase 3: D = A(16x128) x E(128x128) column-tile via v_wmma_f32_16x16x32_f16
    v8f acc = {0.f,0.f,0.f,0.f,0.f,0.f,0.f,0.f};
    const _Float16* arow = &A16[j0 * ASTR];
    const _Float16* brow = &ET16[(nbase + j0) * ESTR];
#pragma unroll
    for (int kc = 0; kc < NTAG; kc += 32) {
      v16h af = load_frag(arow, kc + khalf);
      v16h bf = load_frag(brow, kc + khalf);
      acc = __builtin_amdgcn_wmma_f32_16x16x32_f16(false, af, false, bf, (short)0, acc,
                                                   false, false);
    }
    // phase 4: branchless masked writeback: next = base + ln2*log2(acc) (acc >= 0.6)
#pragma unroll
    for (int r = 0; r < 8; ++r) {
      float ns = fmaf(LN2, LOG2F(acc[r]), base[r]);
      bool keep = ((mw >> (8 * r)) & 0xffull) != 0;
      scoreS[(mrow + r) * SSTR + nbase + j0] = keep ? ns : oldv[r];
    }
    __syncthreads();
  }

  // final: denom[b] = logsumexp_j(score[b][j] + end[j])
  const float* srow = &scoreS[bb * SSTR + cbase];
  float4 s0 = *(const float4*)(srow);
  float4 s1 = *(const float4*)(srow + 4);
  float4 e0 = *(const float4*)(endT + cbase);
  float4 e1 = *(const float4*)(endT + cbase + 4);
  float vals[8] = {s0.x + e0.x, s0.y + e0.y, s0.z + e0.z, s0.w + e0.w,
                   s1.x + e1.x, s1.y + e1.y, s1.z + e1.z, s1.w + e1.w};
  float m = vals[0];
#pragma unroll
  for (int i = 1; i < 8; ++i) m = fmaxf(m, vals[i]);
  m = swz_max16(m);
  float s = 0.f;
#pragma unroll
  for (int i = 0; i < 8; ++i) s += EXP2F((vals[i] - m) * LOG2E);
  s = swz_sum16(s);
  if (j0 == 0) denom[b0 + bb] = m + LN2 * LOG2F(s);
}

// ---------------- numerator (path score) + final mean ----------------
__global__ void __launch_bounds__(64)
crf_score_kernel(const float* __restrict__ em, const int* __restrict__ tags,
                 const unsigned char* __restrict__ mask, const float* __restrict__ startT,
                 const float* __restrict__ endT, const float* __restrict__ trans,
                 const float* __restrict__ denom, float* __restrict__ out) {
  __shared__ float sdata[BATCH];
  const int b = threadIdx.x;
  int t0 = tags[b];
  float sc = startT[t0] + em[(size_t)b * NTAG + t0];
  int prev = t0;
  int cnt = mask[b] ? 1 : 0;
#pragma unroll 8
  for (int i = 1; i < SEQ; ++i) {
    int t = tags[i * BATCH + b];
    if (mask[i * BATCH + b]) {
      sc += trans[prev * NTAG + t] + em[((size_t)i * BATCH + b) * NTAG + t];
      ++cnt;
    }
    prev = t;
  }
  int last = tags[(cnt - 1) * BATCH + b];
  sc += endT[last];
  sdata[b] = denom[b] - sc;
  __syncthreads();
  if (b == 0) {
    float s = 0.f;
    for (int i = 0; i < BATCH; ++i) s += sdata[i];
    *out = s / (float)BATCH;
  }
}

extern "C" void kernel_launch(void* const* d_in, const int* in_sizes, int n_in,
                              void* d_out, int out_size, void* d_ws, size_t ws_size,
                              hipStream_t stream) {
  const float*         em     = (const float*)d_in[0];          // [S,B,T] f32
  const int*           tags   = (const int*)d_in[1];            // [S,B] i32
  const unsigned char* mask   = (const unsigned char*)d_in[2];  // [S,B] bool (1 byte)
  const float*         startT = (const float*)d_in[3];          // [T]
  const float*         endT   = (const float*)d_in[4];          // [T]
  const float*         trans  = (const float*)d_in[5];          // [T,T]
  float* denom = (float*)d_ws;                                  // 64 floats scratch

  crf_forward_kernel<<<dim3(BATCH / TB), dim3(256), 0, stream>>>(em, mask, startT, endT,
                                                                 trans, denom);
  crf_score_kernel<<<dim3(1), dim3(64), 0, stream>>>(em, tags, mask, startT, endT, trans,
                                                     denom, (float*)d_out);
}
